// LinearFP8_41566693490913
// MI455X (gfx1250) — matrix operand order repaired and checked by reading, not compiled, not run on hardware
//
#include <hip/hip_runtime.h>

// ---------------------------------------------------------------------------
// Fused FP8(E4M3)-dequant GEMM + bias for MI455X (gfx1250, wave32, WMMA).
//   out[T, N] = x[T, K] * decode(W[N, K])^T * scale[N] + bias[N]
//   T = 8192, K = 4096, N = 8192
// bf16 WMMA (V_WMMA_F32_16X16X32_BF16), fp32 accumulation.
// Block tile 128x128, 8 waves (4Mx2N), wave tile 32x64 = 2x4 WMMA tiles.
// Software-pipelined: global->reg fetch overlaps WMMA compute; reg->LDS
// commit after compute; one barrier per K-step (double-buffered LDS).
// ---------------------------------------------------------------------------

typedef __attribute__((ext_vector_type(16))) __bf16 v16bf;
typedef __attribute__((ext_vector_type(8)))  __bf16 v8bf;
typedef __attribute__((ext_vector_type(4)))  __bf16 v4bf;
typedef __attribute__((ext_vector_type(8)))  float  v8f;

#define BM 128          // block tile M (tokens)
#define BN 128          // block tile N (output channels)
#define BK 32           // K step (bf16 WMMA K)
#define LDK (BK + 8)    // padded LDS row stride (80B, 16B-aligned rows)
#define NTHREADS 256    // 8 waves

// Branchless E4M3 -> f32 decode (exact). Magnitude bits go to <<20 so the
// 4-bit exponent lands in the f32 exponent field; *2^120 rebias (127-7).
// FP32 denorm-in hardware handles E4M3 subnormals for free. Single cndmask
// remaps the NaN pattern (0x7F magnitude) to 448 per the reference.
__device__ __forceinline__ float decode_e4m3f(int b) {
    const int mag7 = b & 0x7F;
    float mag = __int_as_float(mag7 << 20) * 0x1p120f;
    mag = (mag7 == 0x7F) ? 448.0f : mag;
    return __uint_as_float(__float_as_uint(mag) | ((unsigned)(b & 0x80) << 24));
}

__device__ __forceinline__ v4bf cvt4_bf16(float4 v) {
    v4bf d = { (__bf16)v.x, (__bf16)v.y, (__bf16)v.z, (__bf16)v.w };
    return d;
}

__global__ __launch_bounds__(NTHREADS)
void linear_fp8_wmma_kernel(const float* __restrict__ x,
                            const int* __restrict__ wq,
                            const float* __restrict__ scales,
                            const float* __restrict__ bias,
                            float* __restrict__ out,
                            int T, int K, int N) {
    __shared__ __bf16 Xs[2][BM][LDK];   // 2 * 128 * 80B = 20 KB
    __shared__ __bf16 Ws[2][BN][LDK];   // 2 * 128 * 80B = 20 KB

    const int tid  = threadIdx.x;
    const int lane = tid & 31;
    const int wave = tid >> 5;

    const int t0 = blockIdx.y * BM;     // token row base
    const int o0 = blockIdx.x * BN;     // output channel base

    // 8 waves in a 4(M) x 2(N) grid; each wave owns a 32x64 sub-tile.
    const int wm0 = (wave >> 1) * 32;
    const int wn0 = (wave & 1) * 64;

    const int lr = lane & 15;           // row-within-16 / column-within-16
    const int kh = lane >> 4;           // K half selector (0/1)

    const int KT = K / BK;              // 128 K iterations

    // ---- per-thread staging descriptors (computed once, stepped by BK) ----
    // X: 128x32 fp32 and W: 128x32 int32 -> 1024 x16B each, 4 per thread.
    const float* xp[4];
    const int*   wp[4];
    int soff[4];
#pragma unroll
    for (int i = 0; i < 4; ++i) {
        int idx = tid + i * NTHREADS;             // 0..1023
        int row = idx >> 3;                       // 0..127
        int c4  = (idx & 7) << 2;                 // 0,4,...,28
        xp[i]   = x  + (size_t)(t0 + row) * K + c4;
        wp[i]   = wq + (size_t)(o0 + row) * K + c4;
        soff[i] = row * LDK + c4;
    }
    // prefetch cursors (one cacheline-ish address per thread per array)
    const float* pfx = x  + (size_t)(t0 + (tid >> 1)) * K + (tid & 1) * 16;
    const int*   pfw = wq + (size_t)(o0 + (tid >> 1)) * K + (tid & 1) * 16;

    // staged-in-flight registers
    float4 xr[4];
    int4   wr[4];

    auto fetch = [&]() {                          // issue loads, no waits
#pragma unroll
        for (int i = 0; i < 4; ++i) { xr[i] = *(const float4*)xp[i]; xp[i] += BK; }
#pragma unroll
        for (int i = 0; i < 4; ++i) { wr[i] = *(const int4*)wp[i];   wp[i] += BK; }
    };
    auto commit = [&](int buf) {                  // convert + LDS store
        __bf16* xb = &Xs[buf][0][0];
        __bf16* wb = &Ws[buf][0][0];
#pragma unroll
        for (int i = 0; i < 4; ++i)
            *(v4bf*)(xb + soff[i]) = cvt4_bf16(xr[i]);
#pragma unroll
        for (int i = 0; i < 4; ++i) {
            float4 f = { decode_e4m3f(wr[i].x), decode_e4m3f(wr[i].y),
                         decode_e4m3f(wr[i].z), decode_e4m3f(wr[i].w) };
            *(v4bf*)(wb + soff[i]) = cvt4_bf16(f);
        }
    };

    v8f acc[2][4] = {};                 // 8 x 16x16 fp32 accumulators

    // prologue: tile 0 into LDS buffer 0
    fetch();
    commit(0);
    __syncthreads();

    for (int kt = 0; kt < KT; ++kt) {
        const int cur = kt & 1;
        const bool more = (kt + 1 < KT);
        if (more) fetch();              // loads overlap the WMMA phase below
        if (kt + 2 < KT) {              // gfx1250 global_prefetch_b8
            pfx += BK; pfw += BK;
            __builtin_prefetch(pfx + BK, 0, 1);
            __builtin_prefetch(pfw + BK, 0, 1);
        }

        // ---- load fragments per ISA 7.12.2 bf16 layouts ----
        union V16 { v16bf v; v8bf h[2]; };
        V16 a[2], b[4];
#pragma unroll
        for (int mi = 0; mi < 2; ++mi) {
            // A 16x32: lanes 0-15 row M=lr hold K[0:8]+K[16:24],
            //          lanes 16-31 same rows hold K[8:16]+K[24:32].
            const __bf16* ar = &Xs[cur][wm0 + mi * 16 + lr][0];
            a[mi].h[0] = *(const v8bf*)(ar + kh * 8);
            a[mi].h[1] = *(const v8bf*)(ar + 16 + kh * 8);
        }
#pragma unroll
        for (int ni = 0; ni < 4; ++ni) {
            // B 32x16: lane column N=lr; lanes 0-15 K[0:8]+K[16:24], 16-31 K[8:16]+K[24:32].
            const __bf16* br = &Ws[cur][wn0 + ni * 16 + lr][0];
            b[ni].h[0] = *(const v8bf*)(br + kh * 8);
            b[ni].h[1] = *(const v8bf*)(br + 16 + kh * 8);
        }

#pragma unroll
        for (int mi = 0; mi < 2; ++mi)
#pragma unroll
            for (int ni = 0; ni < 4; ++ni)
                acc[mi][ni] = __builtin_amdgcn_wmma_f32_16x16x32_bf16(
                    /*neg_a=*/false, a[mi].v,
                    /*neg_b=*/false, b[ni].v,
                    /*c_mod=*/(short)0, acc[mi][ni],
                    /*reuse_a=*/false, /*reuse_b=*/false);

        if (more) commit(cur ^ 1);      // waits for loads AFTER compute
        __syncthreads();
    }

    // ---- epilogue: D layout (VGPR j, lanes 0-15: M=j, N=lane;
    //                lanes 16-31: M=j+8, N=lane-16); scale+bias in fp32 ----
    const int mh = kh << 3;
#pragma unroll
    for (int ni = 0; ni < 4; ++ni) {
        const int col = o0 + wn0 + ni * 16 + lr;
        const float sc = scales[col];
        const float bs = bias[col];
#pragma unroll
        for (int mi = 0; mi < 2; ++mi) {
#pragma unroll
            for (int j = 0; j < 8; ++j) {
                const int row = t0 + wm0 + mi * 16 + mh + j;
                out[(size_t)row * N + col] = acc[mi][ni][j] * sc + bs;
            }
        }
    }
}

extern "C" void kernel_launch(void* const* d_in, const int* in_sizes, int n_in,
                              void* d_out, int out_size, void* d_ws, size_t ws_size,
                              hipStream_t stream) {
    const float* x      = (const float*)d_in[0];   // [T, K] fp32
    const int*   wq     = (const int*)d_in[1];     // [N, K] int32 (raw E4M3)
    const float* scales = (const float*)d_in[2];   // [N]
    const float* bias   = (const float*)d_in[3];   // [N]
    float*       out    = (float*)d_out;           // [T, N]

    const int N = in_sizes[2];                     // 8192 output channels
    const int K = in_sizes[1] / N;                 // 4096
    const int T = in_sizes[0] / K;                 // 8192 tokens

    dim3 grid(N / BN, T / BM);                     // (64, 64)
    dim3 block(NTHREADS);
    linear_fp8_wmma_kernel<<<grid, block, 0, stream>>>(x, wq, scales, bias,
                                                       out, T, K, N);
}
